// DSSLayer_35055523070059
// MI455X (gfx1250) — matrix operand order repaired and checked by reading, not compiled, not run on hardware
//
#include <hip/hip_runtime.h>
#include <hip/hip_bf16.h>

// DSS layer for MI455X (gfx1250, wave32, WMMA).
//   kernel 1 (dss_kgen): per-channel DSS kernel generation (complex softmax over
//                        N=32 modes), K stored as f16 in workspace.
//   kernel 2 (dss_conv): causal Toeplitz convolution as block-Toeplitz matmul
//                        with v_wmma_f32_16x16x32_f16, fused + D*u.
//                        All fragments are contiguous 16B-aligned LDS chunks:
//                        A (=u tiles): 2x ds_load_b128, B (=reversed K copies):
//                        2x ds_load_b128, no conditionals in the loop body.

typedef __attribute__((ext_vector_type(16))) _Float16 v16h;
typedef __attribute__((ext_vector_type(8)))  _Float16 v8h;
typedef __attribute__((ext_vector_type(8)))  float    v8f;

constexpr int kH = 1024;
constexpr int kL = 2048;
constexpr int kN = 32;

// LDS layout (halfs) for dss_conv, dynamic shared memory:
//   sKr : 16 reversed/shifted copies of K, stride 2080   [0, 33280)
//   sU  : 256-half zero front pad + u in f16             [33280, 35584)
//   sKt : staging copy of K (forward order)              [35584, 37632)
constexpr int kKrStride = 2080;                 // 2064 used, padded to 32B mult
constexpr int kOffU     = 16 * kKrStride;       // 33280
constexpr int kOffKt    = kOffU + 256 + kL;     // 35584
constexpr int kSmemHalfs = kOffKt + kL;         // 37632 halfs = 75264 bytes

// ---------------------------------------------------------------------------
// Kernel 1: DSS kernel generation.  One 256-thread block per channel.
// K[l] = sum_n Re( coef_n * exp(c_n*l - z_n) )
// ---------------------------------------------------------------------------
__global__ __launch_bounds__(256) void dss_kgen(
    const float* __restrict__ W,         // (H, N, 2)
    const float* __restrict__ Lam_ri,    // (N, 2)
    const float* __restrict__ log_step,  // (H,)
    _Float16*    __restrict__ Kout)      // (H, L) f16
{
    const int h   = blockIdx.x;
    const int tid = threadIdx.x;
    const int n   = tid & 31;   // mode index
    const int grp = tid >> 5;   // 8 partial-sum groups

    __shared__ float sSr[8][32], sSi[8][32];
    __shared__ float sCr[32], sCi[32], sZr[32], sZi[32];
    __shared__ float sPr[32], sPi[32];

    const float step = __expf(log_step[h]);
    const float lr = Lam_ri[2 * n + 0];
    const float li = Lam_ri[2 * n + 1];
    const float cr = step * lr;
    const float ci = step * li;
    // Re(c*l) monotone in l: argmax at l=0 (cr<=0) or l=L-1 (cr>0).
    const float m  = (cr > 0.f) ? (float)(kL - 1) : 0.f;
    const float zr = cr * m, zi = ci * m;

    float ar = 0.f, ai = 0.f;
    for (int l = grp; l < kL; l += 8) {
        float x = cr * (float)l - zr;
        float y = ci * (float)l - zi;
        float ex = __expf(x);
        float sn, cs; __sincosf(y, &sn, &cs);
        ar += ex * cs; ai += ex * sn;
    }
    sSr[grp][n] = ar; sSi[grp][n] = ai;
    __syncthreads();

    if (tid < 32) {
        float sr = 0.f, si = 0.f;
        #pragma unroll
        for (int g = 0; g < 8; ++g) { sr += sSr[g][tid]; si += sSi[g][tid]; }
        const float den = sr * sr + si * si + 1e-7f;   // |s|^2 + eps
        const float mr = sr / den, mi = -si / den;     // conj(s)/den
        const float wr = W[(h * kN + tid) * 2 + 0];
        const float wi = W[(h * kN + tid) * 2 + 1];
        const float l2 = lr * lr + li * li;            // n == tid here
        const float dr = (wr * lr + wi * li) / l2;     // W/Lambda
        const float di = (wi * lr - wr * li) / l2;
        sPr[tid] = dr * mr - di * mi;
        sPi[tid] = dr * mi + di * mr;
        sCr[tid] = cr; sCi[tid] = ci; sZr[tid] = zr; sZi[tid] = zi;
    }
    __syncthreads();

    for (int l = tid; l < kL; l += 256) {
        const float fl = (float)l;
        float acc = 0.f;
        #pragma unroll 8
        for (int nn = 0; nn < kN; ++nn) {
            float x = sCr[nn] * fl - sZr[nn];
            float y = sCi[nn] * fl - sZi[nn];
            float ex = __expf(x);
            float sn, cs; __sincosf(y, &sn, &cs);
            acc += ex * (sPr[nn] * cs - sPi[nn] * sn);
        }
        Kout[h * kL + l] = (_Float16)acc;
    }
}

// ---------------------------------------------------------------------------
// Kernel 2: y = K * u + D*u via block-Toeplitz WMMA.
//
// Output group per wave: D[a,n] = y[16*(ibase+a) + n], a=0..15 output tiles,
// n=0..15 position in tile.  D = sum over lag pairs (d0,d0+1):
//   A[a,k] = u[16*(ibase + a - d0 - (k>>4)) + (k&15)]      (u tiles)
//   B[k,n] = K[16*(d0 + (k>>4)) + n - (k&15)]              (Toeplitz kernel)
//
// The code's A order (lane=M=a, lanes0-15 K=0-15 / lanes16-31 K=16-31) makes the two
// 8-element halves of A contiguous ascending u-chunks, 16B aligned.
// The code's B order (lane&15=N=n, lanes0-15 K=0-15 / lanes16-31 K=16-31)
// makes the whole 16-element B fragment one contiguous ascending chunk of the
// lane's reversed/shifted K copy: sKr[n][q] = K[2048 + n - q], 32B aligned.
// ---------------------------------------------------------------------------
__global__ __launch_bounds__(256) void dss_conv(
    const float*    __restrict__ u,    // (H, L) f32
    const _Float16* __restrict__ Kk,   // (H, L) f16
    const float*    __restrict__ Dv,   // (H,)
    float*          __restrict__ out)  // (H, L) f32
{
    extern __shared__ __align__(32) _Float16 smem[];
    _Float16* sKr = smem;              // 16 copies, stride kKrStride
    _Float16* sU  = smem + kOffU;      // 256 zero pad + 2048
    _Float16* sKt = smem + kOffKt;     // forward K staging

    const int h    = blockIdx.x;
    const int tid  = threadIdx.x;
    const int lane = tid & 31;
    const int wave = tid >> 5;
    const int Mrow = lane & 15;        // A row a
    const int hi   = lane >> 4;        // lane-half select
    const int ncol = lane & 15;        // B/C column n

    // Stage u (f16, front-padded with zeros) and K (forward) into LDS.
    for (int l = tid; l < kL; l += 256) {
        sKt[l]       = Kk[h * kL + l];
        sU[256 + l]  = (_Float16)u[h * kL + l];
    }
    if (tid < 256) sU[tid] = (_Float16)0.f;
    __syncthreads();

    // Build 16 reversed/shifted K copies: sKr[n][q] = K[2048 + n - q] (0 OOB).
    for (int n = 0; n < 16; ++n) {
        _Float16* dst = sKr + n * kKrStride;
        for (int q = tid; q < kKrStride; q += 256) {
            const int ki = 2048 + n - q;
            dst[q] = (ki >= 0 && ki < kL) ? sKt[ki] : (_Float16)0.f;
        }
    }
    __syncthreads();

    const int ibase = wave * 16;           // first output tile of this wave
    const _Float16* myKr = sKr + ncol * kKrStride;
    v8f acc = {};

    for (int d0 = 0; d0 <= ibase + 15; d0 += 2) {   // wave-uniform trips
        // A: one contiguous 16-half u tile (lag d0+hi, k = 16*hi + s), 16B aligned.
        const int tA = 256 + 16 * (ibase + Mrow - d0 - hi);
        const v8h a0 = *(const v8h*)(sU + tA);        // k&15 = 0..7
        const v8h a1 = *(const v8h*)(sU + tA + 8);    // k&15 = 8..15
        // B: one contiguous 16-half chunk of reversed K copy, 32B aligned.
        const v16h Bv = *(const v16h*)(myKr + (2048 - 16 * (d0 + hi)));
        v16h Av;
        #pragma unroll
        for (int i = 0; i < 8; ++i) { Av[i] = a0[i]; Av[8 + i] = a1[i]; }
        acc = __builtin_amdgcn_wmma_f32_16x16x32_f16(
            /*neg_a=*/false, Av, /*neg_b=*/false, Bv,
            /*c_mod=*/(short)0, acc, /*reuse_a=*/false, /*reuse_b=*/false);
    }

    // Epilogue: C/D layout -> row a = r + 8*hi (output tile), col = n.
    const float Dh = Dv[h];
    #pragma unroll
    for (int r = 0; r < 8; ++r) {
        const int a = r + 8 * hi;
        const int t = 16 * (ibase + a) + ncol;
        out[h * kL + t] = acc[r] + Dh * u[h * kL + t];
    }
}

extern "C" void kernel_launch(void* const* d_in, const int* in_sizes, int n_in,
                              void* d_out, int out_size, void* d_ws, size_t ws_size,
                              hipStream_t stream) {
    const float* u     = (const float*)d_in[0];  // (H, L)
    const float* W     = (const float*)d_in[1];  // (H, N, 2)
    const float* Lam   = (const float*)d_in[2];  // (N, 2)
    const float* lstep = (const float*)d_in[3];  // (H,)
    const float* D     = (const float*)d_in[4];  // (H,)
    float* out = (float*)d_out;

    _Float16* Kbuf = (_Float16*)d_ws;            // H*L f16 = 4 MB scratch

    dss_kgen<<<dim3(kH), dim3(256), 0, stream>>>(W, Lam, lstep, Kbuf);
    dss_conv<<<dim3(kH), dim3(256), (unsigned)(kSmemHalfs * sizeof(_Float16)),
               stream>>>(u, Kbuf, D, out);
}
